// SarvamMoESparseMoeBlock_3341484557058
// MI455X (gfx1250) — matrix operand order repaired and checked by reading, not compiled, not run on hardware
//
#include <hip/hip_runtime.h>
#include <math.h>

#define T_TOK 1024
#define H_DIM 2048
#define E_NUM 16
#define K_TOP 4
#define I_DIM 1024
#define IS_DIM 2048
#define ROUTED_SCALING 2.5f

typedef __attribute__((ext_vector_type(16))) __bf16 v16bf;
typedef __attribute__((ext_vector_type(8)))  __bf16 v8bf;
typedef __attribute__((ext_vector_type(4)))  __bf16 v4bf;
typedef __attribute__((ext_vector_type(2)))  __bf16 v2bf;
typedef __attribute__((ext_vector_type(8)))  float  v8f;

// ---------------- WMMA fragment loaders (CDNA5 16x16x32 bf16 layouts) ----------------
// LDS tiles hold K=64; row/col stride is 64 bf16. `base` points at the 32-K segment.
// A: lanes 0-15: row=lane, elems 0-15 = K0..15
//    lanes 16-31: row=lane-16, elems 0-15 = K16..31
__device__ __forceinline__ v16bf load_a_frag64(const __bf16* base, int lane) {
  int row = lane & 15;
  int kh  = (lane >> 4) * 16;
  const __bf16* p = base + row * 64 + kh;
  v8bf lo = *reinterpret_cast<const v8bf*>(p);
  v8bf hi = *reinterpret_cast<const v8bf*>(p + 8);
  return __builtin_shufflevector(lo, hi, 0,1,2,3,4,5,6,7,8,9,10,11,12,13,14,15);
}

// B staged K-major per column: Blds[col][64].
// lanes 0-15: col=lane, elems = K0..15 ; lanes 16-31: col=lane-16, elems = K16..31
__device__ __forceinline__ v16bf load_b_frag64(const __bf16* base, int lane, int wave) {
  int col = (wave << 4) + (lane & 15);
  int kb  = (lane >> 4) * 16;
  return *reinterpret_cast<const v16bf*>(base + col * 64 + kb);
}

__device__ __forceinline__ float silu(float g) {
  // v_rcp_f32 instead of IEEE division expansion; ample for bf16-grade outputs
  return g * __builtin_amdgcn_rcpf(1.f + __expf(-g));
}

// ---- stage 64(K) x 128(N) fp32 weights -> bf16 LDS [col][64], K-major global rows ----
__device__ __forceinline__ void stage_b_kmajor(const float* __restrict__ src, int ldn,
                                               __bf16* Blds, int tid) {
  int kp = tid >> 3;        // k-pair 0..31  (k = 2*kp, 2*kp+1)
  int nb = (tid & 7) * 16;  // 0..112
  const float* r0 = src + (size_t)(2 * kp) * ldn + nb;
  const float* r1 = r0 + ldn;
  __builtin_prefetch(r0 + (size_t)64 * ldn, 0, 1);   // next K-block (global_prefetch_b8)
  __builtin_prefetch(r1 + (size_t)64 * ldn, 0, 1);
#pragma unroll
  for (int j = 0; j < 16; ++j) {
    v2bf p; p[0] = (__bf16)r0[j]; p[1] = (__bf16)r1[j];
    *reinterpret_cast<v2bf*>(Blds + (nb + j) * 64 + 2 * kp) = p;   // packed b32 store
  }
}

// ---- stage 64(K) x 128(N) fp32 weights, N-major global rows (transposed B) ----
__device__ __forceinline__ void stage_b_nmajor(const float* __restrict__ src, int ldk,
                                               __bf16* Blds, int tid) {
  int n  = tid >> 1;        // 0..127
  int kh = (tid & 1) * 32;  // 0 or 32
  const float* p = src + (size_t)n * ldk + kh;
  __builtin_prefetch(p + 64, 0, 1);                  // next K-block of this row
  __bf16* dst = Blds + n * 64 + kh;
#pragma unroll
  for (int j = 0; j < 16; ++j) {
    v2bf q; q[0] = (__bf16)p[2 * j]; q[1] = (__bf16)p[2 * j + 1];
    *reinterpret_cast<v2bf*>(dst + 2 * j) = q;       // contiguous packed b32 stores
  }
}

// ---------------- 0. zero expert counters ----------------
__global__ void zero_counts_kernel(int* counts) {
  if (threadIdx.x < E_NUM) counts[threadIdx.x] = 0;
}

// ---------------- 1. router: one wave32 per token ----------------
__global__ __launch_bounds__(256) void router_kernel(
    const float* __restrict__ x, const float* __restrict__ gate_w,
    const float* __restrict__ bias,
    int* __restrict__ counts, int* __restrict__ tok_list, float* __restrict__ tok_w) {
  const int wave = threadIdx.x >> 5;
  const int lane = threadIdx.x & 31;
  const int t = blockIdx.x * 8 + wave;
  if (t >= T_TOK) return;

  float acc[E_NUM];
#pragma unroll
  for (int e = 0; e < E_NUM; ++e) acc[e] = 0.f;
  const float* xr = x + (size_t)t * H_DIM;
  for (int h = lane; h < H_DIM; h += 32) {
    float xv = xr[h];
#pragma unroll
    for (int e = 0; e < E_NUM; ++e) acc[e] += xv * gate_w[e * H_DIM + h];
  }
#pragma unroll
  for (int e = 0; e < E_NUM; ++e) {
    float v = acc[e];
    for (int off = 16; off > 0; off >>= 1) v += __shfl_xor(v, off, 32);
    acc[e] = v;
  }
  if (lane == 0) {
    float score[E_NUM], sel[E_NUM];
#pragma unroll
    for (int e = 0; e < E_NUM; ++e) {
      score[e] = __builtin_amdgcn_rcpf(1.f + __expf(-acc[e]));
      sel[e] = score[e] + bias[e];
    }
    int ids[K_TOP]; float w[K_TOP]; float wsum = 0.f;
#pragma unroll
    for (int k = 0; k < K_TOP; ++k) {
      int best = 0; float bv = -1e30f;
#pragma unroll
      for (int e = 0; e < E_NUM; ++e)
        if (sel[e] > bv) { bv = sel[e]; best = e; }
      ids[k] = best; w[k] = score[best]; wsum += w[k];
      sel[best] = -1e30f;
    }
    float inv = 1.f / wsum;
#pragma unroll
    for (int k = 0; k < K_TOP; ++k) {
      int e = ids[k];
      int slot = atomicAdd(&counts[e], 1);
      tok_list[e * T_TOK + slot] = t;
      tok_w[e * T_TOK + slot] = w[k] * inv;
    }
  }
}

// ---------------- 2. routed gate+up (gathered GEMM, SiLU*up epilogue) ----------------
__global__ __launch_bounds__(256) void gateup_kernel(
    const float* __restrict__ x,
    const float* __restrict__ w_gate, const float* __restrict__ w_up,
    const int* __restrict__ counts, const int* __restrict__ tok_list,
    __bf16* __restrict__ hbuf) {
  __shared__ __align__(32) __bf16 Alds[16 * 64];
  __shared__ __align__(32) __bf16 Bg[128 * 64];
  __shared__ __align__(32) __bf16 Bu[128 * 64];
  __shared__ int toks[16];

  const int e = blockIdx.z;
  const int cnt = counts[e];
  const int m0 = blockIdx.y * 16;
  if (m0 >= cnt) return;
  const int n0 = blockIdx.x * 128;
  const int tid = threadIdx.x, lane = tid & 31, wave = tid >> 5;

  if (tid < 16) {
    int slot = m0 + tid;
    if (slot >= cnt) slot = cnt - 1;
    toks[tid] = tok_list[e * T_TOK + slot];
  }
  __syncthreads();

  const float* wg = w_gate + (size_t)e * H_DIM * I_DIM;
  const float* wu = w_up + (size_t)e * H_DIM * I_DIM;

  const int arow = (tid * 4) >> 6;   // A staging: 4 elems/thread
  const int akk  = (tid * 4) & 63;

  v8f accg = {}; v8f accu = {};
  for (int k0 = 0; k0 < H_DIM; k0 += 64) {
    { // stage A tile 16x64 (gathered tokens), f32 -> bf16 packed pairs
      const float* xp = x + (size_t)toks[arow] * H_DIM + k0 + akk;
#pragma unroll
      for (int j = 0; j < 4; j += 2) {
        v2bf p; p[0] = (__bf16)xp[j]; p[1] = (__bf16)xp[j + 1];
        *reinterpret_cast<v2bf*>(Alds + arow * 64 + akk + j) = p;
      }
    }
    stage_b_kmajor(wg + (size_t)k0 * I_DIM + n0, I_DIM, Bg, tid);
    stage_b_kmajor(wu + (size_t)k0 * I_DIM + n0, I_DIM, Bu, tid);
    __syncthreads();
#pragma unroll
    for (int s = 0; s < 2; ++s) {
      v16bf a  = load_a_frag64(Alds + s * 32, lane);
      v16bf bg = load_b_frag64(Bg + s * 32, lane, wave);
      v16bf bu = load_b_frag64(Bu + s * 32, lane, wave);
      accg = __builtin_amdgcn_wmma_f32_16x16x32_bf16(false, a, false, bg, (short)0, accg, false, false);
      accu = __builtin_amdgcn_wmma_f32_16x16x32_bf16(false, a, false, bu, (short)0, accu, false, false);
    }
    __syncthreads();
  }

  const int col = n0 + (wave << 4) + (lane & 15);
  __bf16* hrow = hbuf + (size_t)e * T_TOK * I_DIM + col;
  if (m0 + 16 <= cnt) {
#pragma unroll
    for (int r = 0; r < 8; ++r) {
      int m = r + ((lane >> 4) << 3);
      hrow[(size_t)(m0 + m) * I_DIM] = (__bf16)(silu(accg[r]) * accu[r]);
    }
  } else {
#pragma unroll
    for (int r = 0; r < 8; ++r) {
      int m = r + ((lane >> 4) << 3);
      if (m0 + m < cnt)
        hrow[(size_t)(m0 + m) * I_DIM] = (__bf16)(silu(accg[r]) * accu[r]);
    }
  }
}

// ---------------- 3. shared expert gate_up ----------------
__global__ __launch_bounds__(256) void shared_gu_kernel(
    const float* __restrict__ x, const float* __restrict__ wsgu,
    __bf16* __restrict__ hs) {
  __shared__ __align__(32) __bf16 Alds[16 * 64];
  __shared__ __align__(32) __bf16 Bg[128 * 64];
  __shared__ __align__(32) __bf16 Bu[128 * 64];
  const int m0 = blockIdx.y * 16;
  const int n0 = blockIdx.x * 128;
  const int tid = threadIdx.x, lane = tid & 31, wave = tid >> 5;

  const int arow = (tid * 4) >> 6;
  const int akk  = (tid * 4) & 63;

  v8f accg = {}, accu = {};
  for (int k0 = 0; k0 < H_DIM; k0 += 64) {
    {
      const float* xp = x + (size_t)(m0 + arow) * H_DIM + k0 + akk;
#pragma unroll
      for (int j = 0; j < 4; j += 2) {
        v2bf p; p[0] = (__bf16)xp[j]; p[1] = (__bf16)xp[j + 1];
        *reinterpret_cast<v2bf*>(Alds + arow * 64 + akk + j) = p;
      }
    }
    // wsgu is [2*IS, H]: B[k][n] = wsgu[n][k] (transposed, N-major rows)
    stage_b_nmajor(wsgu + (size_t)n0 * H_DIM + k0, H_DIM, Bg, tid);
    stage_b_nmajor(wsgu + (size_t)(IS_DIM + n0) * H_DIM + k0, H_DIM, Bu, tid);
    __syncthreads();
#pragma unroll
    for (int s = 0; s < 2; ++s) {
      v16bf a  = load_a_frag64(Alds + s * 32, lane);
      v16bf bg = load_b_frag64(Bg + s * 32, lane, wave);
      v16bf bu = load_b_frag64(Bu + s * 32, lane, wave);
      accg = __builtin_amdgcn_wmma_f32_16x16x32_bf16(false, a, false, bg, (short)0, accg, false, false);
      accu = __builtin_amdgcn_wmma_f32_16x16x32_bf16(false, a, false, bu, (short)0, accu, false, false);
    }
    __syncthreads();
  }
  const int col = n0 + (wave << 4) + (lane & 15);
#pragma unroll
  for (int r = 0; r < 8; ++r) {
    int m = r + ((lane >> 4) << 3);
    hs[(size_t)(m0 + m) * IS_DIM + col] = (__bf16)(silu(accg[r]) * accu[r]);
  }
}

// ---------------- 4. shared expert down: writes (initializes) d_out ----------------
__global__ __launch_bounds__(256) void shared_down_kernel(
    const __bf16* __restrict__ hs, const float* __restrict__ wsd,
    float* __restrict__ out) {
  __shared__ __align__(32) __bf16 Alds[16 * 64];
  __shared__ __align__(32) __bf16 Bl[128 * 64];
  const int m0 = blockIdx.y * 16;
  const int n0 = blockIdx.x * 128;
  const int tid = threadIdx.x, lane = tid & 31, wave = tid >> 5;

  const int arow = (tid * 4) >> 6;
  const int akk  = (tid * 4) & 63;

  v8f acc = {};
  for (int k0 = 0; k0 < IS_DIM; k0 += 64) {
    { // A already bf16: aligned v4bf (b64) copy
      const v4bf* hp = reinterpret_cast<const v4bf*>(hs + (size_t)(m0 + arow) * IS_DIM + k0 + akk);
      *reinterpret_cast<v4bf*>(Alds + arow * 64 + akk) = *hp;
    }
    // wsd is [H, IS]: B[k=i][n=h] = wsd[n][k] (transposed, N-major rows)
    stage_b_nmajor(wsd + (size_t)n0 * IS_DIM + k0, IS_DIM, Bl, tid);
    __syncthreads();
#pragma unroll
    for (int s = 0; s < 2; ++s) {
      v16bf a = load_a_frag64(Alds + s * 32, lane);
      v16bf b = load_b_frag64(Bl + s * 32, lane, wave);
      acc = __builtin_amdgcn_wmma_f32_16x16x32_bf16(false, a, false, b, (short)0, acc, false, false);
    }
    __syncthreads();
  }
  const int col = n0 + (wave << 4) + (lane & 15);
#pragma unroll
  for (int r = 0; r < 8; ++r) {
    int m = r + ((lane >> 4) << 3);
    out[(size_t)(m0 + m) * H_DIM + col] = acc[r];
  }
}

// ---------------- 5. routed down: scaled scatter-accumulate into d_out ----------------
__global__ __launch_bounds__(256) void down_kernel(
    const __bf16* __restrict__ hbuf, const float* __restrict__ w_down,
    const int* __restrict__ counts, const int* __restrict__ tok_list,
    const float* __restrict__ tok_w, float* __restrict__ out) {
  __shared__ __align__(32) __bf16 Alds[16 * 64];
  __shared__ __align__(32) __bf16 Bl[128 * 64];
  __shared__ int toks[16];
  __shared__ float wrow[16];

  const int e = blockIdx.z;
  const int cnt = counts[e];
  const int m0 = blockIdx.y * 16;
  if (m0 >= cnt) return;
  const int n0 = blockIdx.x * 128;
  const int tid = threadIdx.x, lane = tid & 31, wave = tid >> 5;

  if (tid < 16) {
    int slot = m0 + tid;
    int cs = slot < cnt ? slot : cnt - 1;
    toks[tid] = tok_list[e * T_TOK + cs];
    wrow[tid] = (slot < cnt) ? ROUTED_SCALING * tok_w[e * T_TOK + slot] : 0.f;
  }
  __syncthreads();

  const __bf16* hb = hbuf + (size_t)e * T_TOK * I_DIM;
  const float* wd = w_down + (size_t)e * I_DIM * H_DIM;

  const int arow = (tid * 4) >> 6;
  const int akk  = (tid * 4) & 63;
  int aslot = m0 + arow; if (aslot >= cnt) aslot = cnt - 1;

  v8f acc = {};
  for (int k0 = 0; k0 < I_DIM; k0 += 64) {
    { // A already bf16: aligned v4bf copy
      const v4bf* hp = reinterpret_cast<const v4bf*>(hb + (size_t)aslot * I_DIM + k0 + akk);
      *reinterpret_cast<v4bf*>(Alds + arow * 64 + akk) = *hp;
    }
    // w_down[e] is [I, H]: K-major global rows
    stage_b_kmajor(wd + (size_t)k0 * H_DIM + n0, H_DIM, Bl, tid);
    __syncthreads();
#pragma unroll
    for (int s = 0; s < 2; ++s) {
      v16bf a = load_a_frag64(Alds + s * 32, lane);
      v16bf b = load_b_frag64(Bl + s * 32, lane, wave);
      acc = __builtin_amdgcn_wmma_f32_16x16x32_bf16(false, a, false, b, (short)0, acc, false, false);
    }
    __syncthreads();
  }
  const int col = n0 + (wave << 4) + (lane & 15);
  if (m0 + 16 <= cnt) {
#pragma unroll
    for (int r = 0; r < 8; ++r) {
      int m = r + ((lane >> 4) << 3);
      atomicAdd(&out[(size_t)toks[m] * H_DIM + col], acc[r] * wrow[m]);
    }
  } else {
#pragma unroll
    for (int r = 0; r < 8; ++r) {
      int m = r + ((lane >> 4) << 3);
      if (m0 + m < cnt)
        atomicAdd(&out[(size_t)toks[m] * H_DIM + col], acc[r] * wrow[m]);
    }
  }
}

// ---------------- host launcher ----------------
extern "C" void kernel_launch(void* const* d_in, const int* in_sizes, int n_in,
                              void* d_out, int out_size, void* d_ws, size_t ws_size,
                              hipStream_t stream) {
  const float* x = (const float*)d_in[0];
  const float* gate_w = (const float*)d_in[1];
  const float* bias = (const float*)d_in[2];
  const float* w_gate = (const float*)d_in[3];
  const float* w_up = (const float*)d_in[4];
  const float* w_down = (const float*)d_in[5];
  const float* wsgu = (const float*)d_in[6];
  const float* wsd = (const float*)d_in[7];
  float* out = (float*)d_out;

  char* ws = (char*)d_ws;
  size_t off = 0;
  int* counts = (int*)(ws + off); off += 256;
  int* tok_list = (int*)(ws + off); off += (size_t)E_NUM * T_TOK * sizeof(int);
  float* tok_wts = (float*)(ws + off); off += (size_t)E_NUM * T_TOK * sizeof(float);
  __bf16* hbuf = (__bf16*)(ws + off); off += (size_t)E_NUM * T_TOK * I_DIM * sizeof(__bf16);
  __bf16* hs = (__bf16*)(ws + off); off += (size_t)T_TOK * IS_DIM * sizeof(__bf16);

  hipLaunchKernelGGL(zero_counts_kernel, dim3(1), dim3(64), 0, stream, counts);
  hipLaunchKernelGGL(router_kernel, dim3(T_TOK / 8), dim3(256), 0, stream,
                     x, gate_w, bias, counts, tok_list, tok_wts);
  hipLaunchKernelGGL(gateup_kernel, dim3(I_DIM / 128, T_TOK / 16, E_NUM), dim3(256), 0, stream,
                     x, w_gate, w_up, counts, tok_list, hbuf);
  hipLaunchKernelGGL(shared_gu_kernel, dim3(IS_DIM / 128, T_TOK / 16), dim3(256), 0, stream,
                     x, wsgu, hs);
  hipLaunchKernelGGL(shared_down_kernel, dim3(H_DIM / 128, T_TOK / 16), dim3(256), 0, stream,
                     hs, wsd, out);
  hipLaunchKernelGGL(down_kernel, dim3(H_DIM / 128, T_TOK / 16, E_NUM), dim3(256), 0, stream,
                     hbuf, w_down, counts, tok_list, tok_wts, out);
}
